// CycleRNN_28011776704614
// MI455X (gfx1250) — matrix operand order repaired and checked by reading, not compiled, not run on hardware
//
#include <hip/hip_runtime.h>
#include <hip/hip_bf16.h>
#include <math.h>

#define T_STEPS 300
#define BATCH   256
#define IN_DIM  512
#define HID     512
#define OUT_DIM 128
#define NLAYERS 2

#define NBLK 64
#define NTHR 256

#define LDA 520  // LDS leading dim in halves: 512 + 8 pad (16B) -> bank-conflict-free rows

#if defined(__gfx1250__) && __has_builtin(__builtin_amdgcn_tensor_load_to_lds)
#define USE_TDM 1
#else
#define USE_TDM 0
#endif

typedef __attribute__((ext_vector_type(16))) _Float16 v16h;
typedef __attribute__((ext_vector_type(8)))  _Float16 v8h;
typedef __attribute__((ext_vector_type(8)))  float    v8f;

// ---------------- workspace layout (bytes) ----------------
constexpr size_t OFF_W1  = 0;
constexpr size_t SZ_W1   = (size_t)HID * IN_DIM * 2;                 // f16
constexpr size_t OFF_WIH = OFF_W1 + SZ_W1;
constexpr size_t SZ_WG   = (size_t)NLAYERS * 4 * HID * HID * 2;      // f16
constexpr size_t OFF_WHH = OFF_WIH + SZ_WG;
constexpr size_t OFF_W2  = OFF_WHH + SZ_WG;
constexpr size_t SZ_W2   = (size_t)OUT_DIM * HID * 2;                // f16
constexpr size_t OFF_ZF  = OFF_W2 + SZ_W2;
constexpr size_t SZ_ZF   = (size_t)BATCH * HID * 2;                  // f16
constexpr size_t OFF_HF  = OFF_ZF + SZ_ZF;
constexpr size_t SZ_HF   = (size_t)NLAYERS * 2 * BATCH * HID * 2;    // f16 double-buffered
constexpr size_t OFF_CF  = OFF_HF + SZ_HF;
constexpr size_t SZ_CF   = (size_t)NLAYERS * BATCH * HID * 4;        // f32
constexpr size_t OFF_YF  = OFF_CF + SZ_CF;
constexpr size_t SZ_YF   = (size_t)BATCH * OUT_DIM * 2;              // f16 feedback
constexpr size_t OFF_BAR = OFF_YF + SZ_YF;                           // 2 x u32

// ---------------- fragment helpers ----------------
__device__ __forceinline__ v16h combine16(v8h lo, v8h hi) {
  v16h r;
#pragma unroll
  for (int i = 0; i < 8; ++i) { r[i] = lo[i]; r[i + 8] = hi[i]; }
  return r;
}

// A fragment (16x32 f16) from an LDS tile (16 rows, LDA leading dim):
// lanes 0-15: M=lane, K={k0..k0+7, k0+16..k0+23}; lanes 16-31: K shifted +8
__device__ __forceinline__ v16h lds_a_frag(const _Float16* L, int k0, int lane) {
  const _Float16* p = L + (lane & 15) * LDA + (k0 + ((lane >> 4) << 3));
  return combine16(*(const v8h*)p, *(const v8h*)(p + 16));
}

// B fragment (32x16 f16) for inp @ W^T where W is (N,K) row-major in global:
__device__ __forceinline__ v16h load_b_frag(const _Float16* W, int ld, int n0,
                                            int k0, int lane) {
  const _Float16* p = W + (size_t)(n0 + (lane & 15)) * ld + (k0 + ((lane >> 4) << 3));
  return combine16(*(const v8h*)p, *(const v8h*)(p + 16));
}

__device__ __forceinline__ v8f wmma16(v16h a, v16h b, v8f c) {
  return __builtin_amdgcn_wmma_f32_16x16x32_f16(false, a, false, b, (short)0, c,
                                                false, false);
}

__device__ __forceinline__ float sigf(float v) { return 1.0f / (1.0f + __expf(-v)); }

// ---------------- tile staging: global (16x512 f16, ld=512) -> LDS (ld=LDA) ----------------
#if USE_TDM
typedef __attribute__((ext_vector_type(4))) unsigned int u32x4;
typedef __attribute__((ext_vector_type(8))) int          i32x8;
typedef __attribute__((ext_vector_type(4))) int          i32x4;

// D# per CDNA5 ISA ch.8: group0 = {flags, lds_addr, global_addr, type=2},
// group1 = {data_size=2B + pad(interval=256DW, amount=4DW), dims, tile 512x16, stride 512}
__device__ __forceinline__ void tdm_load_16x512(const _Float16* gsrc, unsigned lds_off) {
  unsigned long long ga = (unsigned long long)(size_t)gsrc;
  u32x4 g0;
  g0[0] = 1u;                                             // count=1, user mode
  g0[1] = lds_off;                                        // lds_addr (bytes)
  g0[2] = (unsigned)ga;                                   // global_addr[31:0]
  g0[3] = ((unsigned)(ga >> 32) & 0x01FFFFFFu) | (2u << 30); // addr[56:32] | type=2
  i32x8 g1;
  g1[0] = (1 << 16) | (1 << 20) | (7 << 22) | (3 << 25);  // 2B elems, pad 4DW/256DW
  g1[1] = (int)(512u << 16);                              // tensor_dim0[15:0]
  g1[2] = (int)(256u << 16);                              // tensor_dim1[15:0]
  g1[3] = (int)(512u << 16);                              // tile_dim0 = 512
  g1[4] = 16;                                             // tile_dim1 = 16
  g1[5] = 512;                                            // tensor_dim0_stride
  g1[6] = 0;
  g1[7] = 0;
  i32x4 z4 = {0, 0, 0, 0};
#if __clang_major__ >= 23
  i32x8 z8 = {0, 0, 0, 0, 0, 0, 0, 0};
  __builtin_amdgcn_tensor_load_to_lds(g0, g1, z4, z4, z8, 0);
#else
  __builtin_amdgcn_tensor_load_to_lds(g0, g1, z4, z4, 0);
#endif
}
#endif

__device__ __forceinline__ void stage_fallback(_Float16* lds, const _Float16* g, int tid) {
  // 16 rows x 64 chunks of 8 halves
  for (int e = tid; e < 16 * 64; e += NTHR) {
    int r = e >> 6, c = (e & 63) << 3;
    *(v8h*)(lds + r * LDA + c) = *(const v8h*)(g + r * 512 + c);
  }
}

// stage two tiles (or one if g2==nullptr) then make LDS visible to the block
__device__ __forceinline__ void stage_tiles(_Float16* l1, const _Float16* g1s,
                                            _Float16* l2, const _Float16* g2s, int tid) {
#if USE_TDM
  if (tid < 32) {
    tdm_load_16x512(g1s, (unsigned)(size_t)(void*)l1);
    if (g2s) tdm_load_16x512(g2s, (unsigned)(size_t)(void*)l2);
    __builtin_amdgcn_s_wait_tensorcnt(0);
  }
#else
  stage_fallback(l1, g1s, tid);
  if (g2s) stage_fallback(l2, g2s, tid);
#endif
  __syncthreads();
}

__device__ __forceinline__ void grid_barrier(unsigned* bar) {
  __threadfence();
  __syncthreads();
  if (threadIdx.x == 0) {
    unsigned* cnt = bar;
    unsigned* gen = bar + 1;
    unsigned g = __hip_atomic_load(gen, __ATOMIC_RELAXED, __HIP_MEMORY_SCOPE_AGENT);
    unsigned a = __hip_atomic_fetch_add(cnt, 1u, __ATOMIC_ACQ_REL, __HIP_MEMORY_SCOPE_AGENT);
    if (a == (unsigned)(NBLK - 1)) {
      __hip_atomic_store(cnt, 0u, __ATOMIC_RELAXED, __HIP_MEMORY_SCOPE_AGENT);
      __hip_atomic_fetch_add(gen, 1u, __ATOMIC_RELEASE, __HIP_MEMORY_SCOPE_AGENT);
    } else {
      while (__hip_atomic_load(gen, __ATOMIC_ACQUIRE, __HIP_MEMORY_SCOPE_AGENT) == g) {
        __builtin_amdgcn_s_sleep(2);
      }
    }
  }
  __syncthreads();
}

// ---------------- prep: f32 -> f16 weights, init state ----------------
__global__ void prep_kernel(const float* __restrict__ w1, const float* __restrict__ wih,
                            const float* __restrict__ whh, const float* __restrict__ w2,
                            const float* __restrict__ x, char* __restrict__ ws) {
  _Float16* w1h  = (_Float16*)(ws + OFF_W1);
  _Float16* wihh = (_Float16*)(ws + OFF_WIH);
  _Float16* whhh = (_Float16*)(ws + OFF_WHH);
  _Float16* w2h  = (_Float16*)(ws + OFF_W2);
  _Float16* hf   = (_Float16*)(ws + OFF_HF);
  float*    cf   = (float*)(ws + OFF_CF);
  _Float16* yf   = (_Float16*)(ws + OFF_YF);
  unsigned* bar  = (unsigned*)(ws + OFF_BAR);

  size_t tid = (size_t)blockIdx.x * blockDim.x + threadIdx.x;
  size_t nt  = (size_t)gridDim.x * blockDim.x;

  for (size_t i = tid; i < (size_t)HID * IN_DIM; i += nt) w1h[i] = (_Float16)w1[i];
  for (size_t i = tid; i < (size_t)NLAYERS * 4 * HID * HID; i += nt) {
    wihh[i] = (_Float16)wih[i];
    whhh[i] = (_Float16)whh[i];
  }
  for (size_t i = tid; i < (size_t)OUT_DIM * HID; i += nt) w2h[i] = (_Float16)w2[i];
  for (size_t i = tid; i < (size_t)NLAYERS * 2 * BATCH * HID; i += nt) hf[i] = (_Float16)0.0f;
  for (size_t i = tid; i < (size_t)NLAYERS * BATCH * HID; i += nt) cf[i] = 0.0f;
  for (size_t i = tid; i < (size_t)BATCH * OUT_DIM; i += nt) {
    size_t b = i >> 7, o = i & (OUT_DIM - 1);
    yf[i] = (_Float16)x[b * IN_DIM + (IN_DIM - OUT_DIM) + o];
  }
  if (tid < 2) bar[tid] = 0u;
}

// ---------------- persistent RNN kernel ----------------
__global__ __launch_bounds__(NTHR, 1) void rnn_persistent(
    const float* __restrict__ x, const float* __restrict__ b1,
    const float* __restrict__ bih, const float* __restrict__ bhh,
    const float* __restrict__ b2, float* __restrict__ out, char* __restrict__ ws) {
  const _Float16* w1h  = (const _Float16*)(ws + OFF_W1);
  const _Float16* wihh = (const _Float16*)(ws + OFF_WIH);
  const _Float16* whhh = (const _Float16*)(ws + OFF_WHH);
  const _Float16* w2h  = (const _Float16*)(ws + OFF_W2);
  _Float16* zf = (_Float16*)(ws + OFF_ZF);
  _Float16* hbuf = (_Float16*)(ws + OFF_HF);
  float*    cf = (float*)(ws + OFF_CF);
  _Float16* yf = (_Float16*)(ws + OFF_YF);
  unsigned* bar = (unsigned*)(ws + OFF_BAR);

  __shared__ _Float16 lsIN[16 * LDA];  // input-activation tile (A operand)
  __shared__ _Float16 lsH[16 * LDA];   // hidden-state tile (A operand)

  const int tid  = threadIdx.x;
  const int lane = tid & 31;
  const int blk  = blockIdx.x;
  const int gwave = blk * (NTHR / 32) + (tid >> 5); // 0..511

  for (int t = 0; t < T_STEPS; ++t) {
    const int pprev = t & 1;
    const int pnew  = pprev ^ 1;
    const int mt    = gwave >> 5;         // shared by all waves of a block
    const int row0  = mt * 16;

    // ---- phase 1: build xt tile in LDS, z = relu(xt @ w1^T + b1) ----
    for (int e = tid; e < 16 * 512; e += NTHR) {
      int r = e >> 9, k = e & (IN_DIM - 1);
      int bb = row0 + r;
      _Float16 v;
      if (k < IN_DIM - OUT_DIM)
        v = (_Float16)x[(size_t)t * BATCH * IN_DIM + (size_t)bb * IN_DIM + k];
      else
        v = yf[bb * OUT_DIM + (k - (IN_DIM - OUT_DIM))];
      lsIN[r * LDA + k] = v;
    }
    __syncthreads();
    {
      const int nt2 = gwave & 31;
      const int col0 = nt2 * 16;
      v8f acc;
#pragma unroll
      for (int i = 0; i < 8; ++i) acc[i] = 0.0f;
      for (int k = 0; k < IN_DIM; k += 32) {
        v16h a = lds_a_frag(lsIN, k, lane);
        v16h w = load_b_frag(w1h, IN_DIM, col0, k, lane);
        acc = wmma16(a, w, acc);
      }
      const int n = lane & 15, mb = (lane >> 4) << 3;
      const int col = col0 + n;
      const float bb = b1[col];
#pragma unroll
      for (int v = 0; v < 8; ++v) {
        float zv = fmaxf(acc[v] + bb, 0.0f);
        zf[(size_t)(row0 + mb + v) * HID + col] = (_Float16)zv;
      }
    }
    grid_barrier(bar);

    // ---- phases 2,3: LSTM layers; wave owns i/f/g/o tiles of one (m,h) block ----
    const _Float16* inp_g = zf;
    for (int l = 0; l < NLAYERS; ++l) {
      const _Float16* wihl = wihh + (size_t)l * 4 * HID * HID;
      const _Float16* whhl = whhh + (size_t)l * 4 * HID * HID;
      const _Float16* hp = hbuf + (size_t)(l * 2 + pprev) * BATCH * HID;
      _Float16*       hn = hbuf + (size_t)(l * 2 + pnew) * BATCH * HID;
      float*          cl = cf + (size_t)l * BATCH * HID;

      // TDM: stage this block's A tiles (input rows + h_prev rows) into LDS
      stage_tiles(lsIN, inp_g + (size_t)row0 * HID, lsH, hp + (size_t)row0 * HID, tid);

      const int hbk = gwave & 31;
      const int hc0 = hbk * 16;
      v8f acc[4];
#pragma unroll
      for (int g = 0; g < 4; ++g)
#pragma unroll
        for (int i = 0; i < 8; ++i) acc[g][i] = 0.0f;

      for (int k = 0; k < HID; k += 32) {
        v16h a  = lds_a_frag(lsIN, k, lane);
        v16h ah = lds_a_frag(lsH, k, lane);
#pragma unroll
        for (int g = 0; g < 4; ++g) {
          v16h bi = load_b_frag(wihl, HID, g * HID + hc0, k, lane);
          acc[g] = wmma16(a, bi, acc[g]);
          v16h bh = load_b_frag(whhl, HID, g * HID + hc0, k, lane);
          acc[g] = wmma16(ah, bh, acc[g]);
        }
      }
      // pointwise LSTM cell in-register
      const int n = lane & 15, mb = (lane >> 4) << 3;
      const int col = hc0 + n;
      const float* bihl = bih + (size_t)l * 4 * HID;
      const float* bhhl = bhh + (size_t)l * 4 * HID;
      const float bi_ = bihl[col] + bhhl[col];
      const float bf_ = bihl[HID + col] + bhhl[HID + col];
      const float bg_ = bihl[2 * HID + col] + bhhl[2 * HID + col];
      const float bo_ = bihl[3 * HID + col] + bhhl[3 * HID + col];
#pragma unroll
      for (int v = 0; v < 8; ++v) {
        const int m = row0 + mb + v;
        float iv = acc[0][v] + bi_;
        float fv = acc[1][v] + bf_;
        float gv = acc[2][v] + bg_;
        float ov = acc[3][v] + bo_;
        float cp = cl[(size_t)m * HID + col];
        float cn = sigf(fv) * cp + sigf(iv) * tanhf(gv);
        float hv = sigf(ov) * tanhf(cn);
        cl[(size_t)m * HID + col] = cn;
        hn[(size_t)m * HID + col] = (_Float16)hv;
      }
      grid_barrier(bar);
      inp_g = hn;
    }

    // ---- phase 4: y = h1 @ w2^T + b2 (blocks 0..15, 8 col-tiles each) ----
    if (blk < (BATCH / 16)) {
      const int r0 = blk * 16;  // blk's own 16-row tile; differs from row0 = (blk>>2)*16
      stage_tiles(lsIN, inp_g + (size_t)r0 * HID, nullptr, nullptr, tid);
      const int ot = gwave & 7;
      const int col0 = ot * 16;
      v8f acc;
#pragma unroll
      for (int i = 0; i < 8; ++i) acc[i] = 0.0f;
      for (int k = 0; k < HID; k += 32) {
        v16h a = lds_a_frag(lsIN, k, lane);
        v16h w = load_b_frag(w2h, HID, col0, k, lane);
        acc = wmma16(a, w, acc);
      }
      const int n = lane & 15, mb = (lane >> 4) << 3;
      const int col = col0 + n;
      const float bb = b2[col];
#pragma unroll
      for (int v = 0; v < 8; ++v) {
        const int m = r0 + mb + v;
        float yv = acc[v] + bb;
        out[(size_t)t * BATCH * OUT_DIM + (size_t)m * OUT_DIM + col] = yv;
        yf[m * OUT_DIM + col] = (_Float16)yv;
      }
    }
    grid_barrier(bar);
  }
}

extern "C" void kernel_launch(void* const* d_in, const int* in_sizes, int n_in,
                              void* d_out, int out_size, void* d_ws, size_t ws_size,
                              hipStream_t stream) {
  const float* x   = (const float*)d_in[0];
  const float* w1  = (const float*)d_in[1];
  const float* b1  = (const float*)d_in[2];
  const float* wih = (const float*)d_in[3];
  const float* whh = (const float*)d_in[4];
  const float* bih = (const float*)d_in[5];
  const float* bhh = (const float*)d_in[6];
  const float* w2  = (const float*)d_in[7];
  const float* b2  = (const float*)d_in[8];
  float* out = (float*)d_out;
  char* ws = (char*)d_ws;

  prep_kernel<<<512, 256, 0, stream>>>(w1, wih, whh, w2, x, ws);
  rnn_persistent<<<NBLK, NTHR, 0, stream>>>(x, b1, bih, bhh, b2, out, ws);
}
